// TransformerEncoderUnit_45956150067410
// MI455X (gfx1250) — compile-verified, matrix operand order checked by reading, not run on hardware
//
#include <hip/hip_runtime.h>

// ---------------- problem constants ----------------
#define B_   2
#define H_   8
#define D_   512
#define DH_  64
#define S_   4096
#define FF_  2048
#define BS_  (B_ * S_)   // 8192 rows

// ---------------- WMMA types ----------------
typedef __attribute__((ext_vector_type(16))) __bf16 v16bf;
typedef __attribute__((ext_vector_type(8)))  float  v8f;
typedef __attribute__((ext_vector_type(4)))  unsigned int u32x4;

union Frag {
    v16bf v;
    u32x4 q[2];
    unsigned short h[16];
};

__device__ __forceinline__ v8f wmma_bf16(v16bf a, v16bf b, v8f c) {
    // D = A(16x32 bf16) * B(32x16 bf16) + C(16x16 f32)
    return __builtin_amdgcn_wmma_f32_16x16x32_bf16(
        false, a, false, b, (short)0, c, false, false);
}

__device__ __forceinline__ void load_frag(Frag& f, const unsigned short* p) {
    // ISA 16-bit fragment: e0..7 = K+0..7, e8..15 = K+16..23 (per half-wave)
    f.q[0] = *(const u32x4*)(p);
    f.q[1] = *(const u32x4*)(p + 16);
}

__device__ __forceinline__ unsigned short f2bf(float f) {
    unsigned int u = __float_as_uint(f);
    u += 0x7FFFu + ((u >> 16) & 1u);   // round-to-nearest-even
    return (unsigned short)(u >> 16);
}

__device__ __forceinline__ float gelu_exact(float x) {
    return 0.5f * x * (1.0f + erff(x * 0.70710678118654752f));
}

__device__ __forceinline__ float rmax16(float v) {
#pragma unroll
    for (int m = 1; m < 16; m <<= 1) v = fmaxf(v, __shfl_xor(v, m, 32));
    return v;
}
__device__ __forceinline__ float rsum16(float v) {
#pragma unroll
    for (int m = 1; m < 16; m <<= 1) v += __shfl_xor(v, m, 32);
    return v;
}

// ---------------- weight prep: fp32 [K,N] -> bf16 [N,K] (K = 1<<kshift) ------
__global__ void __launch_bounds__(256, 1)
transpose_bf16(const float* __restrict__ W,
               unsigned short* __restrict__ Wt,
               int kshift, int N) {
    int t = blockIdx.x * blockDim.x + threadIdx.x;
    if (t >= (N << kshift)) return;
    int n = t >> kshift;
    int k = t & ((1 << kshift) - 1);
    Wt[t] = f2bf(W[(size_t)k * N + n]);
}

// ---------------- layernorm: fp32 [rows,512] -> bf16 ----------------
__global__ void __launch_bounds__(256, 1)
ln_bf16(const float* __restrict__ x,
        const float* __restrict__ g,
        const float* __restrict__ bb,
        unsigned short* __restrict__ out) {
    int row = blockIdx.x;
    int t = threadIdx.x;                 // 256 threads, 2 elems each
    const float* xr = x + (size_t)row * D_;
    float v0 = xr[t], v1 = xr[t + 256];
    __shared__ float red[256];
    red[t] = v0 + v1;
    __syncthreads();
#pragma unroll
    for (int s = 128; s > 0; s >>= 1) {
        if (t < s) red[t] += red[t + s];
        __syncthreads();
    }
    float mean = red[0] * (1.0f / D_);
    __syncthreads();
    float d0 = v0 - mean, d1 = v1 - mean;
    red[t] = d0 * d0 + d1 * d1;
    __syncthreads();
#pragma unroll
    for (int s = 128; s > 0; s >>= 1) {
        if (t < s) red[t] += red[t + s];
        __syncthreads();
    }
    float inv = rsqrtf(red[0] * (1.0f / D_) + 1e-5f);
    out[(size_t)row * D_ + t]       = f2bf(d0 * inv * g[t]       + bb[t]);
    out[(size_t)row * D_ + t + 256] = f2bf(d1 * inv * g[t + 256] + bb[t + 256]);
}

// ---------------- bf16 WMMA GEMM, 32x64 per wave, ping-pong pipelined --------
// C[M,N] = act( A[M,K] * Bt[N,K]^T * outScale + bias ) + resid
// Wave = 2x4 grid of 16x16 WMMA tiles (8 accumulators, 8 WMMA / 12 b128 loads
// per 32-deep K-step). Block(256) = 8 waves as 2x4 -> 64 x 256 output tile.
// Two statically named fragment buffer sets (a0/b0, a1/b1) alternate over a
// K-step of 64 so every array index is a compile-time constant.
// __launch_bounds__(256,1) gives the backend the full VGPR budget -> no spills.
// Requires Kd % 64 == 0 (holds: 512, 2048).
__global__ void __launch_bounds__(256, 1)
gemm_wmma(const unsigned short* __restrict__ A,
          const unsigned short* __restrict__ Bt,
          int M, int N, int Kd,
          const float* __restrict__ bias, int act,
          const float* __restrict__ resid,
          float* __restrict__ outF,
          unsigned short* __restrict__ outB,
          int vtrans, float outScale) {
    const int lane = threadIdx.x & 31;
    const int wid  = threadIdx.x >> 5;
    const int hl   = lane >> 4;     // half-wave selects K sub-block
    const int idx  = lane & 15;     // row (A) / col (B) within 16x16 tile
    const int row0 = blockIdx.y * 64  + (wid >> 2) * 32;
    const int col0 = blockIdx.x * 256 + (wid & 3) * 64;

    const unsigned short* ap[2];
    const unsigned short* bp[4];
    ap[0] = A + (size_t)(row0 + idx) * Kd + hl * 8;
    ap[1] = A + (size_t)(row0 + 16 + idx) * Kd + hl * 8;
#pragma unroll
    for (int j = 0; j < 4; j++)
        bp[j] = Bt + (size_t)(col0 + j * 16 + idx) * Kd + hl * 8;

    Frag a0[2], b0[4], a1[2], b1[4];
#pragma unroll
    for (int i = 0; i < 2; i++) load_frag(a0[i], ap[i]);
#pragma unroll
    for (int j = 0; j < 4; j++) load_frag(b0[j], bp[j]);

    v8f acc[2][4];
#pragma unroll
    for (int i = 0; i < 2; i++)
#pragma unroll
        for (int j = 0; j < 4; j++)
            acc[i][j] = (v8f){0.f,0.f,0.f,0.f,0.f,0.f,0.f,0.f};

    for (int k0 = 0; k0 < Kd; k0 += 64) {
        // issue loads for the odd 32-step while even fragments are consumed
#pragma unroll
        for (int i = 0; i < 2; i++) load_frag(a1[i], ap[i] + k0 + 32);
#pragma unroll
        for (int j = 0; j < 4; j++) load_frag(b1[j], bp[j] + k0 + 32);
        __builtin_prefetch(ap[0] + k0 + 64, 0, 1);
        __builtin_prefetch(bp[0] + k0 + 64, 0, 1);
#pragma unroll
        for (int i = 0; i < 2; i++)
#pragma unroll
            for (int j = 0; j < 4; j++)
                acc[i][j] = wmma_bf16(a0[i].v, b0[j].v, acc[i][j]);
        // issue loads for the next even 32-step while odd fragments are consumed
        if (k0 + 64 < Kd) {
#pragma unroll
            for (int i = 0; i < 2; i++) load_frag(a0[i], ap[i] + k0 + 64);
#pragma unroll
            for (int j = 0; j < 4; j++) load_frag(b0[j], bp[j] + k0 + 64);
        }
#pragma unroll
        for (int i = 0; i < 2; i++)
#pragma unroll
            for (int j = 0; j < 4; j++)
                acc[i][j] = wmma_bf16(a1[i].v, b1[j].v, acc[i][j]);
    }

#pragma unroll
    for (int i = 0; i < 2; i++) {
#pragma unroll
        for (int j = 0; j < 4; j++) {
#pragma unroll
            for (int r = 0; r < 8; r++) {
                int row = row0 + i * 16 + r + 8 * hl;  // C layout: VGPR r + half
                int col = col0 + j * 16 + idx;
                float v = acc[i][j][r] * outScale;
                if (bias)  v += bias[col];
                if (act)   v = gelu_exact(v);
                if (resid) v += resid[(size_t)row * N + col];
                if (outF) {
                    outF[(size_t)row * N + col] = v;
                } else if (vtrans) {
                    // V^T layout [B,H,DH,S]: row = b*S+s, col = h*DH+dh
                    int b  = row >> 12;          // row / S_
                    int s  = row & (S_ - 1);
                    int h  = col >> 6;           // col / DH_
                    int dh = col & (DH_ - 1);
                    outB[(((size_t)b * H_ + h) * DH_ + dh) * S_ + s] = f2bf(v);
                } else {
                    outB[(size_t)row * N + col] = f2bf(v);
                }
            }
        }
    }
}

// ---------------- flash attention ----------------
// Q pre-scaled by 1/sqrt(DH). Q/K bf16 [B,S,H*DH]; V bf16 transposed [B,H,DH,S];
// ctx bf16 [B,S,H*DH]. grid = (B*H, S/128); block = 256 (8 waves x 16 q-rows).
__global__ void __launch_bounds__(256, 1)
flash_attn(const unsigned short* __restrict__ Q,
           const unsigned short* __restrict__ Kg,
           const unsigned short* __restrict__ Vt,
           const int* __restrict__ mask,
           unsigned short* __restrict__ ctx) {
    const int bh = blockIdx.x;
    const int b = bh / H_;
    const int h = bh - b * H_;
    const int wid  = threadIdx.x >> 5;
    const int lane = threadIdx.x & 31;
    const int hl   = lane >> 4;
    const int idx  = lane & 15;
    const int q0   = blockIdx.y * 128 + wid * 16;

    const unsigned short* qg = Q  + ((size_t)b * S_) * D_ + h * DH_;
    const unsigned short* kg = Kg + ((size_t)b * S_) * D_ + h * DH_;
    const unsigned short* vg = Vt + ((size_t)(b * H_ + h) * DH_) * S_;
    unsigned short*       cg = ctx + ((size_t)b * S_) * D_ + h * DH_;

    // Q fragments: 16 rows x 64 dh = two 16x32 A-fragments, loaded once
    Frag qa0, qa1;
    {
        const unsigned short* qp = qg + (size_t)(q0 + idx) * D_ + hl * 8;
        load_frag(qa0, qp);
        load_frag(qa1, qp + 32);
    }

    v8f acc[4];
#pragma unroll
    for (int t = 0; t < 4; t++) acc[t] = (v8f){0.f,0.f,0.f,0.f,0.f,0.f,0.f,0.f};
    float mrow[8], lrow[8];
#pragma unroll
    for (int r = 0; r < 8; r++) { mrow[r] = -3.0e38f; lrow[r] = 0.f; }

    __shared__ unsigned short Plds[8][16 * 32];    // per-wave P transpose buffer
    unsigned short* pl = &Plds[wid][0];

    for (int j = 0; j < S_; j += 32) {
        // ---- issue all K and V fragment loads up front; V loads stay in
        //      flight through the softmax VALU work ----
        Frag kb[4], vb[4];
        {
            const unsigned short* kp0 = kg + (size_t)(j + idx) * D_ + hl * 8;
            const unsigned short* kp1 = kg + (size_t)(j + 16 + idx) * D_ + hl * 8;
            load_frag(kb[0], kp0);          // keys j..j+15,  dh 0..31
            load_frag(kb[1], kp0 + 32);     // keys j..j+15,  dh 32..63
            load_frag(kb[2], kp1);          // keys j+16..31, dh 0..31
            load_frag(kb[3], kp1 + 32);     // keys j+16..31, dh 32..63
        }
#pragma unroll
        for (int t = 0; t < 4; t++) {
            // B-fragment for PV: col = dh (t*16+idx), K = key -> contiguous in V^T
            const unsigned short* vp = vg + (size_t)(t * 16 + idx) * S_ + j + hl * 8;
            load_frag(vb[t], vp);
        }

        // ---- scores: S[16 x 32] = Q(16x64) * K(j..j+31, 64)^T ----
        v8f s0 = {0.f,0.f,0.f,0.f,0.f,0.f,0.f,0.f};
        v8f s1 = {0.f,0.f,0.f,0.f,0.f,0.f,0.f,0.f};
        s0 = wmma_bf16(qa0.v, kb[0].v, s0);
        s0 = wmma_bf16(qa1.v, kb[1].v, s0);
        s1 = wmma_bf16(qa0.v, kb[2].v, s1);
        s1 = wmma_bf16(qa1.v, kb[3].v, s1);

        const int mk0 = mask[(size_t)b * S_ + j + idx];
        const int mk1 = mask[(size_t)b * S_ + j + 16 + idx];

        // ---- online softmax + stash P (bf16) into LDS, transposed ----
#pragma unroll
        for (int r = 0; r < 8; r++) {
            float a0 = mk0 ? s0[r] : -3.0e38f;
            float a1 = mk1 ? s1[r] : -3.0e38f;
            float tmax = rmax16(fmaxf(a0, a1));
            float mn = fmaxf(mrow[r], tmax);
            float sc = __expf(mrow[r] - mn);
            float e0 = __expf(a0 - mn);
            float e1 = __expf(a1 - mn);
            lrow[r] = lrow[r] * sc + rsum16(e0 + e1);
            mrow[r] = mn;
#pragma unroll
            for (int t = 0; t < 4; t++) acc[t][r] *= sc;
            int m = r + 8 * hl;                    // score-tile row owned here
            pl[m * 32 + idx]      = f2bf(e0);
            pl[m * 32 + 16 + idx] = f2bf(e1);
        }
        // wave-internal LDS write->read dependency (cross-lane)
        asm volatile("s_wait_dscnt 0" ::: "memory");

        // P as A-fragment: 16 x 32 (rows = q, K = key index)
        Frag pa;
        load_frag(pa, pl + idx * 32 + hl * 8);

        // ---- acc += P(16x32) * V(32x64) ----
#pragma unroll
        for (int t = 0; t < 4; t++)
            acc[t] = wmma_bf16(pa.v, vb[t].v, acc[t]);
    }

    // ---- normalize and write ctx (bf16) ----
#pragma unroll
    for (int r = 0; r < 8; r++) {
        float inv = (lrow[r] > 0.f) ? (1.0f / lrow[r]) : 0.f;
        int row = q0 + r + 8 * hl;
#pragma unroll
        for (int t = 0; t < 4; t++)
            cg[(size_t)row * D_ + t * 16 + idx] = f2bf(acc[t][r] * inv);
    }
}

// ---------------- workspace layout (bytes; all 256B-aligned) ----------------
static constexpr size_t OFF_XLN = 0;                                  // bf16 8192x512
static constexpr size_t OFF_Q   = OFF_XLN + (size_t)BS_ * D_ * 2;
static constexpr size_t OFF_K   = OFF_Q   + (size_t)BS_ * D_ * 2;
static constexpr size_t OFF_V   = OFF_K   + (size_t)BS_ * D_ * 2;     // V^T [B,H,DH,S]
static constexpr size_t OFF_CTX = OFF_V   + (size_t)BS_ * D_ * 2;
static constexpr size_t OFF_X   = OFF_CTX + (size_t)BS_ * D_ * 2;     // fp32 residual
static constexpr size_t OFF_YLN = OFF_X   + (size_t)BS_ * D_ * 4;
static constexpr size_t OFF_H   = OFF_YLN + (size_t)BS_ * D_ * 2;     // bf16 8192x2048
static constexpr size_t OFF_WQT = OFF_H   + (size_t)BS_ * FF_ * 2;
static constexpr size_t OFF_WKT = OFF_WQT + (size_t)D_ * D_ * 2;
static constexpr size_t OFF_WVT = OFF_WKT + (size_t)D_ * D_ * 2;
static constexpr size_t OFF_WOT = OFF_WVT + (size_t)D_ * D_ * 2;
static constexpr size_t OFF_W1T = OFF_WOT + (size_t)D_ * D_ * 2;      // [FF,D]
static constexpr size_t OFF_W2T = OFF_W1T + (size_t)FF_ * D_ * 2;     // [D,FF]

extern "C" void kernel_launch(void* const* d_in, const int* in_sizes, int n_in,
                              void* d_out, int out_size, void* d_ws, size_t ws_size,
                              hipStream_t stream) {
    const float* reaction = (const float*)d_in[0];
    const int*   mask     = (const int*)d_in[1];
    const float* Wq = (const float*)d_in[2];
    const float* Wk = (const float*)d_in[3];
    const float* Wv = (const float*)d_in[4];
    const float* Wo = (const float*)d_in[5];
    const float* bo = (const float*)d_in[6];
    const float* W1 = (const float*)d_in[7];
    const float* b1 = (const float*)d_in[8];
    const float* W2 = (const float*)d_in[9];
    const float* b2 = (const float*)d_in[10];
    const float* g_sa = (const float*)d_in[11];
    const float* b_sa = (const float*)d_in[12];
    const float* g_ff = (const float*)d_in[13];
    const float* b_ff = (const float*)d_in[14];

    char* ws = (char*)d_ws;
    unsigned short* xln = (unsigned short*)(ws + OFF_XLN);
    unsigned short* Qb  = (unsigned short*)(ws + OFF_Q);
    unsigned short* Kb  = (unsigned short*)(ws + OFF_K);
    unsigned short* Vtb = (unsigned short*)(ws + OFF_V);
    unsigned short* ctx = (unsigned short*)(ws + OFF_CTX);
    float*          xbf = (float*)(ws + OFF_X);
    unsigned short* yln = (unsigned short*)(ws + OFF_YLN);
    unsigned short* hbf = (unsigned short*)(ws + OFF_H);
    unsigned short* wqT = (unsigned short*)(ws + OFF_WQT);
    unsigned short* wkT = (unsigned short*)(ws + OFF_WKT);
    unsigned short* wvT = (unsigned short*)(ws + OFF_WVT);
    unsigned short* woT = (unsigned short*)(ws + OFF_WOT);
    unsigned short* w1T = (unsigned short*)(ws + OFF_W1T);
    unsigned short* w2T = (unsigned short*)(ws + OFF_W2T);

    // 1) weights -> bf16 transposed [out,in]  (K is a power of two -> shift)
    {
        int nDD = D_ * D_, nDF = D_ * FF_;
        transpose_bf16<<<(nDD + 255) / 256, 256, 0, stream>>>(Wq, wqT, 9,  D_);
        transpose_bf16<<<(nDD + 255) / 256, 256, 0, stream>>>(Wk, wkT, 9,  D_);
        transpose_bf16<<<(nDD + 255) / 256, 256, 0, stream>>>(Wv, wvT, 9,  D_);
        transpose_bf16<<<(nDD + 255) / 256, 256, 0, stream>>>(Wo, woT, 9,  D_);
        transpose_bf16<<<(nDF + 255) / 256, 256, 0, stream>>>(W1, w1T, 9,  FF_);
        transpose_bf16<<<(nDF + 255) / 256, 256, 0, stream>>>(W2, w2T, 11, D_);
    }

    // 2) LN1 -> bf16
    ln_bf16<<<BS_, 256, 0, stream>>>(reaction, g_sa, b_sa, xln);

    // 3) Q/K/V projections (Q pre-scaled by 1/sqrt(DH)=0.125; V written transposed)
    dim3 gDD(D_ / 256, BS_ / 64);     // (2, 128)
    gemm_wmma<<<gDD, 256, 0, stream>>>(xln, wqT, BS_, D_, D_, nullptr, 0, nullptr,
                                       nullptr, Qb, 0, 0.125f);
    gemm_wmma<<<gDD, 256, 0, stream>>>(xln, wkT, BS_, D_, D_, nullptr, 0, nullptr,
                                       nullptr, Kb, 0, 1.0f);
    gemm_wmma<<<gDD, 256, 0, stream>>>(xln, wvT, BS_, D_, D_, nullptr, 0, nullptr,
                                       nullptr, Vtb, 1, 1.0f);

    // 4) flash attention -> ctx (bf16)
    flash_attn<<<dim3(B_ * H_, S_ / 128), 256, 0, stream>>>(Qb, Kb, Vtb, mask, ctx);

    // 5) output proj + GELU + residual -> x (fp32)
    gemm_wmma<<<gDD, 256, 0, stream>>>(ctx, woT, BS_, D_, D_, bo, 1, reaction,
                                       xbf, nullptr, 0, 1.0f);

    // 6) LN2 -> bf16
    ln_bf16<<<BS_, 256, 0, stream>>>(xbf, g_ff, b_ff, yln);

    // 7) FFN up + GELU -> h (bf16)
    dim3 gDF(FF_ / 256, BS_ / 64);    // (8, 128)
    gemm_wmma<<<gDF, 256, 0, stream>>>(yln, w1T, BS_, FF_, D_, b1, 1, nullptr,
                                       nullptr, hbf, 0, 1.0f);

    // 8) FFN down + residual -> out (fp32)
    gemm_wmma<<<gDD, 256, 0, stream>>>(hbf, w2T, BS_, D_, FF_, b2, 0, xbf,
                                       (float*)d_out, nullptr, 0, 1.0f);
}
